// PointTransformerLayer_67439576482093
// MI455X (gfx1250) — matrix operand order repaired and checked by reading, not compiled, not run on hardware
//
#include <hip/hip_runtime.h>
#include <hip/hip_bf16.h>

typedef _Float16 v16h __attribute__((ext_vector_type(16)));
typedef _Float16 v8h  __attribute__((ext_vector_type(8)));
typedef float    v8f  __attribute__((ext_vector_type(8)));

#define NSEG 32
#define SEG  2048
#define NPT  (NSEG*SEG)   // 65536
#define C    256
#define CS   32
#define NS   16
#define QC   768          // q|k|v fused columns

static __device__ __forceinline__ v16h cat16(v8h lo, v8h hi) {
  v16h r;
  #pragma unroll
  for (int i = 0; i < 8; ++i) { r[i] = lo[i]; r[i + 8] = hi[i]; }
  return r;
}
static __device__ __forceinline__ v8h ld8h(const _Float16* p) {
  return *reinterpret_cast<const v8h*>(p);
}

// ---------------------------------------------------------------------------
// Kernel 1: cast x -> f16, build transposed f16 weights for WMMA B-fragments
// ---------------------------------------------------------------------------
__global__ void prep_kernel(const float* __restrict__ x,
                            const float* __restrict__ Wq, const float* __restrict__ Wk,
                            const float* __restrict__ Wv,
                            const float* __restrict__ Ww1, const float* __restrict__ Ww2,
                            _Float16* __restrict__ xh, _Float16* __restrict__ wqkvt,
                            _Float16* __restrict__ ww1t, _Float16* __restrict__ ww2t)
{
  const int XTOT = NPT * C;
  const int WQKV = 3 * C * C;
  const int W1   = CS * C;
  const int W2   = CS * CS;
  const int total = XTOT + WQKV + W1 + W2;
  for (int i = blockIdx.x * blockDim.x + threadIdx.x; i < total;
       i += gridDim.x * blockDim.x) {
    if (i < XTOT) {
      xh[i] = (_Float16)x[i];
    } else if (i < XTOT + WQKV) {
      int e = i - XTOT;
      int w = e / (C * C);
      int r = e % (C * C);
      int k = r & (C - 1);
      int nl = r >> 8;
      const float* W = (w == 0) ? Wq : (w == 1) ? Wk : Wv;
      wqkvt[(w * C + nl) * C + k] = (_Float16)W[k * C + nl];   // [col][k]
    } else if (i < XTOT + WQKV + W1) {
      int e = i - XTOT - WQKV;
      int k = e & (C - 1);
      int nl = e >> 8;
      ww1t[nl * C + k] = (_Float16)Ww1[k * CS + nl];
    } else {
      int e = i - XTOT - WQKV - W1;
      int k = e & (CS - 1);
      int nl = e >> 5;
      ww2t[nl * CS + k] = (_Float16)Ww2[k * CS + nl];
    }
  }
}

// ---------------------------------------------------------------------------
// Kernel 2: fused QKV GEMM  [N x 256] * [256 x 768] -> qkv f16 (L2-resident)
// Each wave computes a 16x64 strip: one A fragment feeds 4 WMMAs (4x A reuse).
// ---------------------------------------------------------------------------
__global__ __launch_bounds__(256) void qkv_kernel(
    const _Float16* __restrict__ xh, const _Float16* __restrict__ wt,
    const float* __restrict__ bq, const float* __restrict__ bk,
    const float* __restrict__ bv, _Float16* __restrict__ qkv)
{
  int wave = threadIdx.x >> 5, lane = threadIdx.x & 31;
  int job = blockIdx.x * 8 + wave;            // 4096 * 12 wave-jobs
  int tm = job / (QC / 64);
  int tg = job % (QC / 64);                   // 64-column group
  int l16 = lane & 15, hf = lane >> 4;
  int arow = tm * 16 + l16;
  const _Float16* ap = xh + arow * C;
  const _Float16* bp = wt + (tg * 64 + l16) * C;   // +u*16*C per sub-tile
  v8f acc[4] = {};
  #pragma unroll
  for (int kk = 0; kk < C; kk += 32) {
    v16h af = cat16(ld8h(ap + kk + hf * 8), ld8h(ap + kk + 16 + hf * 8));
    #pragma unroll
    for (int u = 0; u < 4; ++u) {
      const _Float16* bu = bp + u * 16 * C + kk + hf * 8;
      v16h bf = cat16(ld8h(bu), ld8h(bu + 16));
      acc[u] = __builtin_amdgcn_wmma_f32_16x16x32_f16(false, af, false, bf,
                                                      (short)0, acc[u], false, false);
    }
  }
  #pragma unroll
  for (int u = 0; u < 4; ++u) {
    int bcol = tg * 64 + u * 16 + l16;
    float bias = (bcol < C) ? bq[bcol] : (bcol < 2 * C) ? bk[bcol - C] : bv[bcol - 2 * C];
    #pragma unroll
    for (int r = 0; r < 8; ++r) {
      int orow = tm * 16 + r + 8 * hf;        // D layout: row = r + 8*(lane>=16)
      qkv[orow * QC + bcol] = (_Float16)(acc[u][r] + bias);
    }
  }
}

// ---------------------------------------------------------------------------
// Kernel 3: brute-force kNN (top-16) within each 2048-point segment.
// Segment staged in LDS via CDNA5 async global->LDS copies (ASYNCcnt).
// ---------------------------------------------------------------------------
__global__ __launch_bounds__(256) void knn_kernel(const float* __restrict__ p,
                                                  int* __restrict__ nidx)
{
  __shared__ float s_p[SEG * 3];              // 24 KB, [point][xyz]
  int seg = blockIdx.x >> 3, part = blockIdx.x & 7;
  int base = seg * SEG;
  const float* gsrc = p + (size_t)base * 3;
  // async copy: 6144 floats = 1536 x b128; 6 per thread, no divergence
  #pragma unroll
  for (int k = 0; k < (SEG * 3) / (4 * 256); ++k) {
    int i = (threadIdx.x + k * 256) * 4;
    unsigned ldsa = (unsigned)(uintptr_t)&s_p[i];        // LDS aperture: low 32 bits
    const float* ga = gsrc + i;
    asm volatile("global_load_async_to_lds_b128 %0, %1, off"
                 :: "v"(ldsa), "v"(ga) : "memory");
  }
  asm volatile("s_wait_asynccnt 0x0" ::: "memory");
  __syncthreads();

  int q = part * 256 + threadIdx.x;
  float qx = s_p[q * 3 + 0], qy = s_p[q * 3 + 1], qz = s_p[q * 3 + 2];
  float bd[NS]; int bi[NS];
  #pragma unroll
  for (int s = 0; s < NS; ++s) { bd[s] = 3.4e38f; bi[s] = 0; }
  for (int j = 0; j < SEG; ++j) {
    float dx = s_p[j * 3 + 0] - qx;
    float dy = s_p[j * 3 + 1] - qy;
    float dz = s_p[j * 3 + 2] - qz;
    float d2 = dx * dx + dy * dy + dz * dz;
    if (d2 < bd[NS - 1]) {
      float d = d2; int id = j;
      #pragma unroll
      for (int s = 0; s < NS; ++s) {
        if (d < bd[s]) {
          float td = bd[s]; int ti = bi[s];
          bd[s] = d; bi[s] = id; d = td; id = ti;
        }
      }
    }
  }
  int gq = base + q;
  #pragma unroll
  for (int s = 0; s < NS; ++s) nidx[gq * NS + s] = base + bi[s];
}

// ---------------------------------------------------------------------------
// Kernel 4: fused gather + pos-encoding + linear_w (WMMA) + softmax + output.
// One wave per point; channel mapping c = lane*8+j so every gather is a b128.
// ---------------------------------------------------------------------------
#define AW 4
__global__ __launch_bounds__(AW * 32) void attn_kernel(
    const float* __restrict__ p, const _Float16* __restrict__ qkv,
    const int* __restrict__ nidx,
    const float* __restrict__ Wp1, const float* __restrict__ bp1,
    const float* __restrict__ gp, const float* __restrict__ bpln,
    const float* __restrict__ Wp2, const float* __restrict__ bp2,
    const float* __restrict__ gw1, const float* __restrict__ bw1ln,
    const _Float16* __restrict__ ww1t, const float* __restrict__ bw1,
    const float* __restrict__ gw2, const float* __restrict__ bw2ln,
    const _Float16* __restrict__ ww2t, const float* __restrict__ bw2,
    float* __restrict__ out)
{
  __shared__ _Float16 s_ww2[CS * CS];     // Ww2^T, B-frag friendly
  __shared__ float s_wp2[4 * C];          // Wp2 rows 0..2, row 3 = bp2
  __shared__ float s_gb1[2 * C];          // gw1, bw1ln
  __shared__ float s_sm[4 * CS];          // bw1 | gw2 | bw2ln | bw2
  __shared__ float s_pw[20];              // Wp1(9) bp1(3) gp(3) bpln(3)
  __shared__ int      s_idx[AW][NS];
  __shared__ float    s_a[AW][NS][4];     // post-LN3 relu'd p features
  __shared__ _Float16 s_rn[AW][NS][C];    // normalized r (WMMA A-tile source)
  __shared__ _Float16 s_u[AW][NS][CS];    // normalized u1
  __shared__ float    s_w[AW][NS][CS];    // softmax weights

  int tid = threadIdx.x;
  int wave = tid >> 5, lane = tid & 31;
  int l16 = lane & 15, hf = lane >> 4;
  int n = blockIdx.x * AW + wave;

  // --- stage 0: block-cooperative staging of small weights -----------------
  for (int i = tid; i < CS * CS; i += AW * 32) s_ww2[i] = ww2t[i];
  for (int i = tid; i < C; i += AW * 32) {
    s_wp2[i]         = Wp2[0 * C + i];
    s_wp2[C + i]     = Wp2[1 * C + i];
    s_wp2[2 * C + i] = Wp2[2 * C + i];
    s_wp2[3 * C + i] = bp2[i];
    s_gb1[i]         = gw1[i];
    s_gb1[C + i]     = bw1ln[i];
  }
  if (tid < CS) {
    s_sm[tid]          = bw1[tid];
    s_sm[CS + tid]     = gw2[tid];
    s_sm[2 * CS + tid] = bw2ln[tid];
    s_sm[3 * CS + tid] = bw2[tid];
  }
  if (tid < 9) s_pw[tid] = Wp1[tid];
  if (tid < 3) { s_pw[9 + tid] = bp1[tid]; s_pw[12 + tid] = gp[tid]; s_pw[15 + tid] = bpln[tid]; }
  if (lane < NS) s_idx[wave][lane] = nidx[n * NS + lane];
  __syncthreads();

  // --- stage p: tiny 3-dim pipeline, lanes 0..15 handle neighbor t=lane ----
  if (lane < NS) {
    int g = s_idx[wave][lane];
    float prx = p[g * 3 + 0] - p[n * 3 + 0];
    float pry = p[g * 3 + 1] - p[n * 3 + 1];
    float prz = p[g * 3 + 2] - p[n * 3 + 2];
    float pp[3];
    #pragma unroll
    for (int d = 0; d < 3; ++d)
      pp[d] = prx * s_pw[0 * 3 + d] + pry * s_pw[1 * 3 + d] + prz * s_pw[2 * 3 + d] + s_pw[9 + d];
    float m = (pp[0] + pp[1] + pp[2]) * (1.0f / 3.0f);
    float v = ((pp[0] - m) * (pp[0] - m) + (pp[1] - m) * (pp[1] - m) +
               (pp[2] - m) * (pp[2] - m)) * (1.0f / 3.0f);
    float inv = rsqrtf(v + 1e-5f);
    #pragma unroll
    for (int d = 0; d < 3; ++d) {
      float a = (pp[d] - m) * inv * s_pw[12 + d] + s_pw[15 + d];
      s_a[wave][lane][d] = fmaxf(a, 0.0f);
    }
  }
  __syncthreads();

  // hoisted t-invariant per-channel constants, c = lane*8 + j
  float w0[8], w1[8], w2[8], w3[8], g1v[8], be1v[8];
  #pragma unroll
  for (int j = 0; j < 8; ++j) {
    int c = lane * 8 + j;
    w0[j] = s_wp2[c];       w1[j] = s_wp2[C + c];
    w2[j] = s_wp2[2*C + c]; w3[j] = s_wp2[3*C + c];
    g1v[j] = s_gb1[c];      be1v[j] = s_gb1[C + c];
  }

  // --- stage r: gather xk (b128/lane), r = xk - xq + pr, LN256+ReLU -> LDS -
  v8h xqv = ld8h(qkv + n * QC + lane * 8);
  float xq8[8];
  #pragma unroll
  for (int j = 0; j < 8; ++j) xq8[j] = (float)xqv[j];
  for (int t = 0; t < NS; ++t) {
    int g = s_idx[wave][t];
    if (t + 1 < NS) __builtin_prefetch(qkv + s_idx[wave][t + 1] * QC + C, 0, 1);
    float a0 = s_a[wave][t][0], a1 = s_a[wave][t][1], a2 = s_a[wave][t][2];
    v8h kv = ld8h(qkv + g * QC + C + lane * 8);
    float rv[8]; float sum = 0.f, ssq = 0.f;
    #pragma unroll
    for (int j = 0; j < 8; ++j) {
      float pr = a0 * w0[j] + a1 * w1[j] + a2 * w2[j] + w3[j];
      float v = (float)kv[j] - xq8[j] + pr;
      rv[j] = v; sum += v; ssq += v * v;
    }
    #pragma unroll
    for (int msk = 1; msk < 32; msk <<= 1) {
      sum += __shfl_xor(sum, msk, 32);
      ssq += __shfl_xor(ssq, msk, 32);
    }
    float mean = sum * (1.0f / 256.0f);
    float var  = ssq * (1.0f / 256.0f) - mean * mean;
    float inv = rsqrtf(var + 1e-5f);
    v8h rn;
    #pragma unroll
    for (int j = 0; j < 8; ++j)
      rn[j] = (_Float16)fmaxf((rv[j] - mean) * inv * g1v[j] + be1v[j], 0.0f);
    *reinterpret_cast<v8h*>(&s_rn[wave][t][lane * 8]) = rn;   // single b128 DS store
  }
  __syncthreads();

  // --- stage WMMA1: u1 = rn[16x256] @ Ww1[256x32] -> two 16x16 tiles -------
  v8f acc0 = {}, acc1 = {};
  {
    const _Float16* aB = &s_rn[wave][l16][0];
    #pragma unroll
    for (int kk = 0; kk < C; kk += 32) {
      v16h af = cat16(ld8h(aB + kk + hf * 8), ld8h(aB + kk + 16 + hf * 8));
      const _Float16* b0p = ww1t + l16 * C + kk + hf * 8;
      const _Float16* b1p = ww1t + (16 + l16) * C + kk + hf * 8;
      v16h b0 = cat16(ld8h(b0p), ld8h(b0p + 16));
      v16h b1 = cat16(ld8h(b1p), ld8h(b1p + 16));
      acc0 = __builtin_amdgcn_wmma_f32_16x16x32_f16(false, af, false, b0,
                                                    (short)0, acc0, false, false);
      acc1 = __builtin_amdgcn_wmma_f32_16x16x32_f16(false, af, false, b1,
                                                    (short)0, acc1, false, false);
    }
  }
  // --- u1: +bw1, LN over 32 channels (half-wave butterfly), ReLU -> s_u ----
  {
    float b0  = s_sm[l16],          b1  = s_sm[16 + l16];
    float g0  = s_sm[CS + l16],     g1  = s_sm[CS + 16 + l16];
    float be0 = s_sm[2 * CS + l16], be1 = s_sm[2 * CS + 16 + l16];
    #pragma unroll
    for (int r = 0; r < 8; ++r) {
      float u0 = acc0[r] + b0;
      float u1 = acc1[r] + b1;
      float s = u0 + u1, q = u0 * u0 + u1 * u1;
      #pragma unroll
      for (int msk = 1; msk < 16; msk <<= 1) {   // stays within lane half
        s += __shfl_xor(s, msk, 32);
        q += __shfl_xor(q, msk, 32);
      }
      float mean = s * (1.0f / 32.0f);
      float var  = q * (1.0f / 32.0f) - mean * mean;
      float inv = rsqrtf(var + 1e-5f);
      int row = r + 8 * hf;                      // D layout row
      s_u[wave][row][l16]      = (_Float16)fmaxf((u0 - mean) * inv * g0 + be0, 0.0f);
      s_u[wave][row][16 + l16] = (_Float16)fmaxf((u1 - mean) * inv * g1 + be1, 0.0f);
    }
  }
  __syncthreads();

  // --- stage WMMA2: u2 = u1n[16x32] @ Ww2[32x32] ---------------------------
  v8f c0 = {}, c1 = {};
  {
    const _Float16* a2p = &s_u[wave][l16][0];
    v16h a2 = cat16(ld8h(a2p + hf * 8), ld8h(a2p + 16 + hf * 8));
    const _Float16* b0p = &s_ww2[l16 * CS + hf * 8];
    const _Float16* b1p = &s_ww2[(16 + l16) * CS + hf * 8];
    v16h b0 = cat16(ld8h(b0p), ld8h(b0p + 16));
    v16h b1 = cat16(ld8h(b1p), ld8h(b1p + 16));
    c0 = __builtin_amdgcn_wmma_f32_16x16x32_f16(false, a2, false, b0,
                                                (short)0, c0, false, false);
    c1 = __builtin_amdgcn_wmma_f32_16x16x32_f16(false, a2, false, b1,
                                                (short)0, c1, false, false);
  }
  // --- softmax over the 16 neighbors (rows) per channel --------------------
  {
    float bb0 = s_sm[3 * CS + l16], bb1 = s_sm[3 * CS + 16 + l16];
    float e0[8], e1[8];
    float m0 = -3.4e38f, m1 = -3.4e38f;
    #pragma unroll
    for (int r = 0; r < 8; ++r) {
      e0[r] = c0[r] + bb0; e1[r] = c1[r] + bb1;
      m0 = fmaxf(m0, e0[r]); m1 = fmaxf(m1, e1[r]);
    }
    m0 = fmaxf(m0, __shfl_xor(m0, 16, 32));      // other 8 rows live in lane^16
    m1 = fmaxf(m1, __shfl_xor(m1, 16, 32));
    float sm0 = 0.f, sm1 = 0.f;
    #pragma unroll
    for (int r = 0; r < 8; ++r) {
      e0[r] = __expf(e0[r] - m0); e1[r] = __expf(e1[r] - m1);
      sm0 += e0[r]; sm1 += e1[r];
    }
    sm0 += __shfl_xor(sm0, 16, 32);
    sm1 += __shfl_xor(sm1, 16, 32);
    float i0 = 1.0f / sm0, i1 = 1.0f / sm1;
    #pragma unroll
    for (int r = 0; r < 8; ++r) {
      int row = r + 8 * hf;
      s_w[wave][row][l16]      = e0[r] * i0;
      s_w[wave][row][16 + l16] = e1[r] * i1;
    }
  }
  __syncthreads();

  // --- aggregate: out[c] = sum_t (xv_g[t,c] + pr[t,c]) * w[t, c%32] --------
  float accv[8];
  #pragma unroll
  for (int j = 0; j < 8; ++j) accv[j] = 0.f;
  for (int t = 0; t < NS; ++t) {
    int g = s_idx[wave][t];
    float a0 = s_a[wave][t][0], a1 = s_a[wave][t][1], a2 = s_a[wave][t][2];
    v8h vv = ld8h(qkv + g * QC + 2 * C + lane * 8);       // b128 gather
    const float* wr = &s_w[wave][t][(lane & 3) * 8];      // w[t, (lane*8+j)%32]
    #pragma unroll
    for (int j = 0; j < 8; ++j) {
      float pr = a0 * w0[j] + a1 * w1[j] + a2 * w2[j] + w3[j];
      accv[j] += ((float)vv[j] + pr) * wr[j];
    }
  }
  float4* o4 = reinterpret_cast<float4*>(out + n * C + lane * 8);
  o4[0] = make_float4(accv[0], accv[1], accv[2], accv[3]);
  o4[1] = make_float4(accv[4], accv[5], accv[6], accv[7]);
}

// ---------------------------------------------------------------------------
extern "C" void kernel_launch(void* const* d_in, const int* in_sizes, int n_in,
                              void* d_out, int out_size, void* d_ws, size_t ws_size,
                              hipStream_t stream) {
  const float* p    = (const float*)d_in[0];
  const float* x    = (const float*)d_in[1];
  const float* Wq   = (const float*)d_in[2];
  const float* bq   = (const float*)d_in[3];
  const float* Wk   = (const float*)d_in[4];
  const float* bk   = (const float*)d_in[5];
  const float* Wv   = (const float*)d_in[6];
  const float* bv   = (const float*)d_in[7];
  const float* Wp1  = (const float*)d_in[8];
  const float* bp1  = (const float*)d_in[9];
  const float* gp   = (const float*)d_in[10];
  const float* bpln = (const float*)d_in[11];
  const float* Wp2  = (const float*)d_in[12];
  const float* bp2  = (const float*)d_in[13];
  const float* gw1  = (const float*)d_in[14];
  const float* bw1ln= (const float*)d_in[15];
  const float* Ww1  = (const float*)d_in[16];
  const float* bw1  = (const float*)d_in[17];
  const float* gw2  = (const float*)d_in[18];
  const float* bw2ln= (const float*)d_in[19];
  const float* Ww2  = (const float*)d_in[20];
  const float* bw2  = (const float*)d_in[21];
  float* out = (float*)d_out;

  char* ws = (char*)d_ws;
  size_t off = 0;
  _Float16* xh    = (_Float16*)(ws + off); off += (size_t)NPT * C * 2;      // 33.5 MB
  _Float16* qkv   = (_Float16*)(ws + off); off += (size_t)NPT * QC * 2;     // 100.7 MB
  _Float16* wqkvt = (_Float16*)(ws + off); off += (size_t)3 * C * C * 2;
  _Float16* ww1t  = (_Float16*)(ws + off); off += (size_t)CS * C * 2;
  _Float16* ww2t  = (_Float16*)(ws + off); off += (size_t)CS * CS * 2;
  off = (off + 255) & ~(size_t)255;
  int* nidx = (int*)(ws + off);            off += (size_t)NPT * NS * 4;     // 4.2 MB

  prep_kernel<<<16384, 256, 0, stream>>>(x, Wq, Wk, Wv, Ww1, Ww2,
                                         xh, wqkvt, ww1t, ww2t);
  qkv_kernel<<<(NPT / 16) * (QC / 64) / 8, 256, 0, stream>>>(xh, wqkvt, bq, bk, bv, qkv);
  knn_kernel<<<NSEG * (SEG / 256), 256, 0, stream>>>(p, nidx);
  attn_kernel<<<NPT / AW, AW * 32, 0, stream>>>(p, qkv, nidx,
                                                Wp1, bp1, gp, bpln, Wp2, bp2,
                                                gw1, bw1ln, ww1t, bw1,
                                                gw2, bw2ln, ww2t, bw2, out);
}
